// NetV2_22943715295879
// MI455X (gfx1250) — compile-verified, matrix operand order checked by reading, not run on hardware
//
#include <hip/hip_runtime.h>
#include <math.h>

// ---------------------------------------------------------------------------
// Sparse CNN on MI455X (gfx1250): batch is the WMMA M dimension.
// bf16 activations/weights, f32 accumulation via v_wmma_f32_16x16x32_bf16.
// ---------------------------------------------------------------------------

typedef __bf16 bf16_t;
typedef __attribute__((ext_vector_type(8)))  __bf16 v8bf;
typedef __attribute__((ext_vector_type(16))) __bf16 v16bf;
typedef __attribute__((ext_vector_type(8)))  float  v8f;

static __device__ inline bf16_t f2bf(float f) {
    unsigned u = __builtin_bit_cast(unsigned, f);
    unsigned r = u + 0x7FFFu + ((u >> 16) & 1u);          // round-to-nearest-even
    unsigned short s = (unsigned short)(r >> 16);
    return __builtin_bit_cast(bf16_t, s);
}
static __device__ inline v16bf cat16(v8bf lo, v8bf hi) {
    return __builtin_shufflevector(lo, hi, 0,1,2,3,4,5,6,7,8,9,10,11,12,13,14,15);
}
// 16-bit operand fragment K index for element e (0..15), lane-half lh (0/1).
static __device__ inline int kmap(int e, int lh) {
    return (e < 8 ? e : 16 + (e - 8)) + 8 * lh;
}

// ---------------- workspace layout (bytes) ----------------
constexpr size_t alup(size_t x) { return (x + 255) & ~(size_t)255; }
constexpr int NP = 150, NB = 2048;
constexpr size_t O_SLOTMAP = 0;                               // 784 int
constexpr size_t O_SY    = alup(O_SLOTMAP + 784*4);           // 150 int
constexpr size_t O_SX    = alup(O_SY + NP*4);                 // 150 int
constexpr size_t O_P2    = alup(O_SX + NP*4);                 // 150*9 int (tap<<16|q)
constexpr size_t O_CNT2  = alup(O_P2 + NP*9*4);               // 150 int
constexpr size_t O_CLIST = alup(O_CNT2 + NP*4);               // 196 int (dense cell id)
constexpr size_t O_CPC   = alup(O_CLIST + 196*4);             // 196 int
constexpr size_t O_CP    = alup(O_CPC + 196*4);               // 196*4 int (tap<<16|j)
constexpr size_t O_NC    = alup(O_CP + 196*4*4);              // 1 int
constexpr size_t O_A1    = alup(O_NC + 4);                    // 32 f
constexpr size_t O_C1    = alup(O_A1 + 32*4);                 // 32 f
constexpr size_t O_A2    = alup(O_C1 + 32*4);                 // 64 f
constexpr size_t O_C2    = alup(O_A2 + 64*4);                 // 64 f
constexpr size_t O_A3    = alup(O_C2 + 64*4);                 // 64 f
constexpr size_t O_C3    = alup(O_A3 + 64*4);                 // 64 f
constexpr size_t O_W1P   = alup(O_C3 + 64*4);                 // 288 f
constexpr size_t O_W2P   = alup(O_W1P + 288*4);               // 18432 bf16 (9 taps, frag order)
constexpr size_t O_W3P   = alup(O_W2P + 18432*2);             // 16384 bf16 (4 taps x 2 ksteps)
constexpr size_t O_FC1P  = alup(O_W3P + 16384*2);             // <= 12544*128 bf16
constexpr size_t O_H1    = alup(O_FC1P + (size_t)12544*128*2);// 2048*150*32 bf16
constexpr size_t O_H2    = alup(O_H1 + (size_t)NB*NP*32*2);   // 2048*150*64 bf16
constexpr size_t O_H3    = alup(O_H2 + (size_t)NB*NP*64*2);   // 2048*(ncell*64) bf16
constexpr size_t O_Z     = alup(O_H3 + (size_t)NB*12544*2);   // 2048*128 f32
// total ~116 MB

#define WSF(o) ((float*)(ws + (o)))
#define WSI(o) ((int*)(ws + (o)))
#define WSB(o) ((bf16_t*)(ws + (o)))

// ---------------- prep: metadata, pair lists, folded BN ----------------
__global__ void k_prep_meta(char* ws, const int* idxs,
    const float* g1, const float* bb1, const float* m1, const float* v1,
    const float* g2, const float* bb2, const float* m2, const float* v2,
    const float* g3, const float* bb3, const float* m3, const float* v3,
    const float* w1)
{
    if (threadIdx.x != 0 || blockIdx.x != 0) return;
    int* slotmap = WSI(O_SLOTMAP);
    int* sy = WSI(O_SY); int* sx = WSI(O_SX);
    for (int i = 0; i < 784; ++i) slotmap[i] = -1;
    for (int j = 0; j < NP; ++j) {                 // rows 0..149 = batch 0, feature order
        int y = idxs[j*3 + 1], x = idxs[j*3 + 2];
        sy[j] = y; sx[j] = x; slotmap[y*28 + x] = j;
    }
    // conv1/conv2 active-neighbor pairs (3x3, pad 1)
    int* p2 = WSI(O_P2); int* cnt2 = WSI(O_CNT2);
    for (int j = 0; j < NP; ++j) {
        int c = 0;
        for (int t = 0; t < 9; ++t) {
            int ny = sy[j] + t/3 - 1, nx = sx[j] + t%3 - 1;
            if (ny < 0 || ny >= 28 || nx < 0 || nx >= 28) continue;
            int q = slotmap[ny*28 + nx];
            if (q >= 0) p2[j*9 + c++] = (t << 16) | q;
        }
        cnt2[j] = c;
    }
    // active pooled cells (2x2 maxpool of mask) + per-cell (tap, j) pairs
    int* clist = WSI(O_CLIST); int* cpc = WSI(O_CPC); int* cp = WSI(O_CP);
    int nc = 0;
    for (int cy = 0; cy < 14; ++cy)
        for (int cx = 0; cx < 14; ++cx) {
            int tmp[4]; int pc = 0;
            for (int t = 0; t < 4; ++t) {
                int fy = cy*2 + (t >> 1), fx = cx*2 + (t & 1);
                int q = slotmap[fy*28 + fx];
                if (q >= 0) tmp[pc++] = (t << 16) | q;
            }
            if (pc > 0) {
                clist[nc] = cy*14 + cx; cpc[nc] = pc;
                for (int i = 0; i < pc; ++i) cp[nc*4 + i] = tmp[i];
                ++nc;
            }
        }
    WSI(O_NC)[0] = nc;
    // fold BN: bn(h) = h*a + c
    float* a1 = WSF(O_A1); float* c1 = WSF(O_C1);
    for (int c = 0; c < 32; ++c) { a1[c] = g1[c] * rsqrtf(v1[c] + 1e-5f); c1[c] = bb1[c] - m1[c]*a1[c]; }
    float* a2 = WSF(O_A2); float* c2 = WSF(O_C2);
    for (int c = 0; c < 64; ++c) { a2[c] = g2[c] * rsqrtf(v2[c] + 1e-5f); c2[c] = bb2[c] - m2[c]*a2[c]; }
    float* a3 = WSF(O_A3); float* c3 = WSF(O_C3);
    for (int c = 0; c < 64; ++c) { a3[c] = g3[c] * rsqrtf(v3[c] + 1e-5f); c3[c] = bb3[c] - m3[c]*a3[c]; }
    float* w1p = WSF(O_W1P);
    for (int i = 0; i < 288; ++i) w1p[i] = w1[i];  // (3,3,1,32) flat == tap*32+c
}

// ---------------- pack w2/w3 into WMMA B-fragment order ----------------
__global__ void k_pack_w(char* ws, const float* w2, const float* w3) {
    int idx = blockIdx.x * 256 + threadIdx.x;
    if (idx < 18432) {                       // w2: [tap(9)][nt(4)][lane(32)][e(16)]
        int e = idx & 15, lane = (idx >> 4) & 31, nt = (idx >> 9) & 3, t = idx >> 11;
        int k = kmap(e, lane >> 4);
        int n = nt*16 + (lane & 15);
        WSB(O_W2P)[idx] = f2bf(w2[((size_t)t*32 + k)*64 + n]);
    } else if (idx < 34816) {                // w3: [tap(4)][kk(2)][nt(4)][lane(32)][e(16)]
        int i2 = idx - 18432;
        int e = i2 & 15, lane = (i2 >> 4) & 31, nt = (i2 >> 9) & 3, kk = (i2 >> 11) & 1, t = i2 >> 12;
        int k = kk*32 + kmap(e, lane >> 4);
        int n = nt*16 + (lane & 15);
        WSB(O_W3P)[i2] = f2bf(w3[((size_t)t*64 + k)*64 + n]);
    }
}

// pack fc1_w active rows: compact k = cellIdx*64 + co -> row co*196 + cell
__global__ void k_pack_fc1(char* ws, const float* fc1w) {
    int nk = WSI(O_NC)[0] * 2;               // ksteps of 32
    long idx = (long)blockIdx.x * 256 + threadIdx.x;
    if (idx >= (long)nk * 4096) return;      // [ks][nt(8)][lane(32)][e(16)]
    int e = (int)(idx & 15), lane = (int)((idx >> 4) & 31), nt = (int)((idx >> 9) & 7);
    long ks = idx >> 12;
    int kc = (int)ks*32 + kmap(e, lane >> 4);
    int ci = kc >> 6, co = kc & 63;
    int cell = WSI(O_CLIST)[ci];
    int row = co*196 + cell;                 // NCHW flatten: c*196 + y*14 + x
    int n = nt*16 + (lane & 15);
    WSB(O_FC1P)[idx] = f2bf(fc1w[(size_t)row*128 + n]);
}

// ---------------- conv1 (Cin=1, scalar) ----------------
__global__ void __launch_bounds__(256) k_conv1(char* ws, const float* feat) {
    int id = blockIdx.x * 256 + threadIdx.x;
    if (id >= NB * NP) return;
    int b = id / NP, j = id - b * NP;
    const int* cnt2 = WSI(O_CNT2); const int* p2 = WSI(O_P2);
    const float* w1p = WSF(O_W1P);
    const float* a1 = WSF(O_A1); const float* c1 = WSF(O_C1);
    float acc[32];
    #pragma unroll
    for (int c = 0; c < 32; ++c) acc[c] = 0.f;
    int cnt = cnt2[j];
    for (int p = 0; p < cnt; ++p) {
        int pr = p2[j*9 + p]; int t = pr >> 16, q = pr & 0xFFFF;
        float f = feat[(size_t)b*NP + q];
        const float* wp = w1p + t*32;
        #pragma unroll
        for (int c = 0; c < 32; ++c) acc[c] += f * wp[c];
    }
    bf16_t* o = WSB(O_H1) + ((size_t)b*NP + j) * 32;
    #pragma unroll
    for (int c = 0; c < 32; ++c) o[c] = f2bf(fmaxf(acc[c]*a1[c] + c1[c], 0.f));
}

// ---------------- conv2: per-tap K=32 WMMA, M=16 batch rows, N=64 ----------------
__global__ void __launch_bounds__(128) k_conv2(char* ws) {
    __shared__ bf16_t sw[18432];                         // all 9 taps, 36 KB
    const bf16_t* w2p = WSB(O_W2P);
    for (int i = threadIdx.x; i < 2304; i += 128)
        ((v8bf*)sw)[i] = ((const v8bf*)w2p)[i];
    __syncthreads();
    const int wave = threadIdx.x >> 5, lane = threadIdx.x & 31;
    const int m = lane & 15, lh = lane >> 4;
    const int b0 = (blockIdx.x * 4 + wave) * 16;
    const int j0 = blockIdx.y * 15;
    const int j1 = (j0 + 15 < NP) ? j0 + 15 : NP;
    const bf16_t* h1 = WSB(O_H1);
    bf16_t* h2 = WSB(O_H2);
    const int* cnt2 = WSI(O_CNT2); const int* p2 = WSI(O_P2);
    const float* a2 = WSF(O_A2); const float* c2 = WSF(O_C2);
    for (int j = j0; j < j1; ++j) {
        v8f acc[4];
        #pragma unroll
        for (int nt = 0; nt < 4; ++nt)
            #pragma unroll
            for (int r = 0; r < 8; ++r) acc[nt][r] = 0.f;
        const int cnt = cnt2[j];
        for (int p = 0; p < cnt; ++p) {
            int pr = p2[j*9 + p]; int t = pr >> 16, q = pr & 0xFFFF;
            const bf16_t* ap = h1 + ((size_t)(b0 + m)*NP + q) * 32 + lh*8;
            v16bf A = cat16(*(const v8bf*)ap, *(const v8bf*)(ap + 16));
            #pragma unroll
            for (int nt = 0; nt < 4; ++nt) {
                const bf16_t* bp = sw + t*2048 + nt*512 + lane*16;
                v16bf Bf = cat16(*(const v8bf*)bp, *(const v8bf*)(bp + 8));
                acc[nt] = __builtin_amdgcn_wmma_f32_16x16x32_bf16(
                    false, A, false, Bf, (short)0, acc[nt], false, false);
            }
        }
        #pragma unroll
        for (int nt = 0; nt < 4; ++nt) {
            int c = nt*16 + m;
            float av = a2[c], cv = c2[c];
            #pragma unroll
            for (int r = 0; r < 8; ++r) {
                float v = fmaxf(acc[nt][r]*av + cv, 0.f);
                h2[((size_t)(b0 + r + 8*lh)*NP + j)*64 + c] = f2bf(v);
            }
        }
    }
}

// ---------------- conv3: K=64 (2 WMMA ksteps), compact active cells ----------------
__global__ void __launch_bounds__(128) k_conv3(char* ws) {
    __shared__ bf16_t sw[16384];                         // 4 taps x 64x64, 32 KB
    const bf16_t* w3p = WSB(O_W3P);
    for (int i = threadIdx.x; i < 2048; i += 128)
        ((v8bf*)sw)[i] = ((const v8bf*)w3p)[i];
    __syncthreads();
    const int wave = threadIdx.x >> 5, lane = threadIdx.x & 31;
    const int m = lane & 15, lh = lane >> 4;
    const int b0 = (blockIdx.x * 4 + wave) * 16;
    const int nc = WSI(O_NC)[0];
    const size_t K = (size_t)nc * 64;
    int ci0 = blockIdx.y * 14;
    int ci1 = (ci0 + 14 < nc) ? ci0 + 14 : nc;
    const bf16_t* h2 = WSB(O_H2);
    bf16_t* h3 = WSB(O_H3);
    const int* cpc = WSI(O_CPC); const int* cp = WSI(O_CP);
    const float* a3 = WSF(O_A3); const float* c3 = WSF(O_C3);
    for (int ci = ci0; ci < ci1; ++ci) {
        v8f acc[4];
        #pragma unroll
        for (int nt = 0; nt < 4; ++nt)
            #pragma unroll
            for (int r = 0; r < 8; ++r) acc[nt][r] = 0.f;
        int pc = cpc[ci];
        for (int p = 0; p < pc; ++p) {
            int pr = cp[ci*4 + p]; int t = pr >> 16, j = pr & 0xFFFF;
            const bf16_t* base = h2 + ((size_t)(b0 + m)*NP + j) * 64;
            v16bf A0 = cat16(*(const v8bf*)(base + lh*8),      *(const v8bf*)(base + 16 + lh*8));
            v16bf A1 = cat16(*(const v8bf*)(base + 32 + lh*8), *(const v8bf*)(base + 48 + lh*8));
            #pragma unroll
            for (int nt = 0; nt < 4; ++nt) {
                const bf16_t* bp0 = sw + t*4096 + nt*512 + lane*16;
                v16bf B0 = cat16(*(const v8bf*)bp0, *(const v8bf*)(bp0 + 8));
                acc[nt] = __builtin_amdgcn_wmma_f32_16x16x32_bf16(
                    false, A0, false, B0, (short)0, acc[nt], false, false);
                const bf16_t* bp1 = bp0 + 2048;
                v16bf B1 = cat16(*(const v8bf*)bp1, *(const v8bf*)(bp1 + 8));
                acc[nt] = __builtin_amdgcn_wmma_f32_16x16x32_bf16(
                    false, A1, false, B1, (short)0, acc[nt], false, false);
            }
        }
        #pragma unroll
        for (int nt = 0; nt < 4; ++nt) {
            int c = nt*16 + m;
            float av = a3[c], cv = c3[c];
            #pragma unroll
            for (int r = 0; r < 8; ++r) {
                float v = fmaxf(acc[nt][r]*av + cv, 0.f);
                h3[(size_t)(b0 + r + 8*lh)*K + (size_t)ci*64 + c] = f2bf(v);
            }
        }
    }
}

// ---------------- fc1: M=16/wave, N=128, compact K (LDS-staged B) ----------------
__global__ void __launch_bounds__(128) k_fc1(char* ws, const float* fc1b) {
    __shared__ bf16_t sb[4096];                          // one kstep of B (32x128), 8 KB
    const int wave = threadIdx.x >> 5, lane = threadIdx.x & 31;
    const int m = lane & 15, lh = lane >> 4;
    const int b0 = (blockIdx.x * 4 + wave) * 16;
    const int nk = WSI(O_NC)[0] * 2;
    const size_t K = (size_t)nk * 32;
    const bf16_t* h3 = WSB(O_H3);
    const bf16_t* fp = WSB(O_FC1P);
    float* z = WSF(O_Z);
    v8f acc[8];
    #pragma unroll
    for (int nt = 0; nt < 8; ++nt)
        #pragma unroll
        for (int r = 0; r < 8; ++r) acc[nt][r] = 0.f;
    for (int ks = 0; ks < nk; ++ks) {
        __syncthreads();
        const v8bf* src = (const v8bf*)(fp + (size_t)ks * 4096);
        for (int i = threadIdx.x; i < 512; i += 128) ((v8bf*)sb)[i] = src[i];
        __syncthreads();
        const bf16_t* ap = h3 + (size_t)(b0 + m)*K + (size_t)ks*32 + lh*8;
        v16bf A = cat16(*(const v8bf*)ap, *(const v8bf*)(ap + 16));
        #pragma unroll
        for (int nt = 0; nt < 8; ++nt) {
            const bf16_t* bp = sb + nt*512 + lane*16;
            v16bf Bf = cat16(*(const v8bf*)bp, *(const v8bf*)(bp + 8));
            acc[nt] = __builtin_amdgcn_wmma_f32_16x16x32_bf16(
                false, A, false, Bf, (short)0, acc[nt], false, false);
        }
    }
    #pragma unroll
    for (int nt = 0; nt < 8; ++nt) {
        int n = nt*16 + m;
        float bias = fc1b[n];
        #pragma unroll
        for (int r = 0; r < 8; ++r)
            z[(size_t)(b0 + r + 8*lh)*128 + n] = fmaxf(acc[nt][r] + bias, 0.f);
    }
}

// ---------------- fc2 + log_softmax (one wave per batch row) ----------------
__global__ void __launch_bounds__(128) k_fc2(char* ws, const float* fw,
                                             const float* fb, float* out) {
    __shared__ float sfw[1280];
    for (int i = threadIdx.x; i < 1280; i += 128) sfw[i] = fw[i];
    __syncthreads();
    const int wave = threadIdx.x >> 5, lane = threadIdx.x & 31;
    const int b = blockIdx.x * 4 + wave;
    const float* zr = WSF(O_Z) + (size_t)b * 128;
    float acc[10];
    #pragma unroll
    for (int j = 0; j < 10; ++j) acc[j] = 0.f;
    #pragma unroll
    for (int kk = 0; kk < 4; ++kk) {
        int k = lane + kk*32;
        float zv = zr[k];
        #pragma unroll
        for (int j = 0; j < 10; ++j) acc[j] += zv * sfw[k*10 + j];
    }
    #pragma unroll
    for (int j = 0; j < 10; ++j)
        for (int off = 16; off > 0; off >>= 1)
            acc[j] += __shfl_xor(acc[j], off, 32);
    if (lane == 0) {
        float lg[10], mx = -1e30f;
        #pragma unroll
        for (int j = 0; j < 10; ++j) { lg[j] = acc[j] + fb[j]; mx = fmaxf(mx, lg[j]); }
        float s = 0.f;
        #pragma unroll
        for (int j = 0; j < 10; ++j) s += expf(lg[j] - mx);
        float lse = mx + logf(s);
        #pragma unroll
        for (int j = 0; j < 10; ++j) out[(size_t)b*10 + j] = lg[j] - lse;
    }
}

// ---------------- host ----------------
extern "C" void kernel_launch(void* const* d_in, const int* in_sizes, int n_in,
                              void* d_out, int out_size, void* d_ws, size_t ws_size,
                              hipStream_t stream) {
    const float* feat = (const float*)d_in[0];
    const int*   idxs = (const int*)d_in[1];
    const float* w1   = (const float*)d_in[3];
    const float* g1 = (const float*)d_in[4],  *bb1 = (const float*)d_in[5];
    const float* m1 = (const float*)d_in[6],  *v1  = (const float*)d_in[7];
    const float* w2   = (const float*)d_in[8];
    const float* g2 = (const float*)d_in[9],  *bb2 = (const float*)d_in[10];
    const float* m2 = (const float*)d_in[11], *v2  = (const float*)d_in[12];
    const float* w3   = (const float*)d_in[13];
    const float* g3 = (const float*)d_in[14], *bb3 = (const float*)d_in[15];
    const float* m3 = (const float*)d_in[16], *v3  = (const float*)d_in[17];
    const float* fc1w = (const float*)d_in[18];
    const float* fc1b = (const float*)d_in[19];
    const float* fc2w = (const float*)d_in[20];
    const float* fc2b = (const float*)d_in[21];
    char* ws = (char*)d_ws;

    k_prep_meta<<<1, 32, 0, stream>>>(ws, idxs, g1, bb1, m1, v1,
                                      g2, bb2, m2, v2, g3, bb3, m3, v3, w1);
    k_pack_w  <<<136, 256, 0, stream>>>(ws, w2, w3);
    k_pack_fc1<<<6272, 256, 0, stream>>>(ws, fc1w);
    k_conv1   <<<1200, 256, 0, stream>>>(ws, feat);
    k_conv2   <<<dim3(32, 10), 128, 0, stream>>>(ws);
    k_conv3   <<<dim3(32, 14), 128, 0, stream>>>(ws);
    k_fc1     <<<32, 128, 0, stream>>>(ws, fc1b);
    k_fc2     <<<512, 128, 0, stream>>>(ws, fc2w, fc2b, (float*)d_out);
}
